// Ernie4_5_VLDecoderLayer_25933012533748
// MI455X (gfx1250) — compile-verified, matrix operand order checked by reading, not run on hardware
//
#include <hip/hip_runtime.h>
#include <math.h>

// ---------------- problem constants ----------------
#define NTOK 1024
#define HDIM 1024
#define NH   8
#define NKV  2
#define HD   128
#define NEXP 16
#define KTOP 6
#define IDIM 512
#define SIDIM 1024
#define QKVW ((NH + 2*NKV) * HD)   // 1536
#define HALF 64
#define TFREQ 20
#define HWCUT (HALF - TFREQ)       // 44

// ---------------- WMMA types ----------------
typedef __attribute__((ext_vector_type(16))) __bf16 v16bf;
typedef __attribute__((ext_vector_type(8)))  __bf16 v8bf;
typedef __attribute__((ext_vector_type(8)))  float  v8f;

// GEMM block tiling: 256(M) x 32(N), K-step 64. 8 waves = 4(M) x 2(N),
// 4 accumulators per wave (one B fragment feeds 4 WMMAs). Double-buffered
// LDS staging: one barrier per K-tile, next-tile global loads overlap WMMAs.
#define GBM 256
#define GBN 32
#define GBK 64
#define GSTR 72   // padded LDS col stride (bf16): 144B, 16B-aligned

// attention-score tiling (no weights; keep small tiles)
#define STM 32
#define STN 64
#define STK 32

__device__ __forceinline__ __bf16 f2bf(float f) { return (__bf16)f; }

// Load a 16-element bf16 A/B fragment from a row-major row pointer.
// Per ISA 7.12.2 (16-bit A 16x32): lane&15 selects row/col, lane>>4 selects
// K-half; elements 0..7 = K[kbase..kbase+7], 8..15 = K[kbase+16..kbase+23].
__device__ __forceinline__ v16bf load_frag16(const __bf16* rowptr, int kbase) {
  v8bf lo = *(const v8bf*)(rowptr + kbase);
  v8bf hi = *(const v8bf*)(rowptr + kbase + 16);
  v16bf out;
#pragma unroll
  for (int i = 0; i < 8; ++i) { out[i] = lo[i]; out[i + 8] = hi[i]; }
  return out;
}

// Build a fragment from LDS stored fragment-major: sBT[col*GSTR + k].
__device__ __forceinline__ v16bf frag_from_lds(const __bf16* bp) {
  v8bf lo = *(const v8bf*)bp;          // K = khalf .. khalf+7
  v8bf hi = *(const v8bf*)(bp + 16);   // K = khalf+16 .. khalf+23
  v16bf out;
#pragma unroll
  for (int i = 0; i < 8; ++i) { out[i] = lo[i]; out[i + 8] = hi[i]; }
  return out;
}

__device__ __forceinline__ v8f wmma_bf16(v16bf a, v16bf b, v8f c) {
  return __builtin_amdgcn_wmma_f32_16x16x32_bf16(false, a, false, b,
                                                 (short)0, c, false, false);
}

// ================= RMSNorm -> bf16 =================
__global__ void __launch_bounds__(256)
rmsnorm_bf16(const float* __restrict__ X, const float* __restrict__ W,
             __bf16* __restrict__ Out, int Hdim) {
  int n = blockIdx.x;
  const float* row = X + (size_t)n * Hdim;
  float ss = 0.f;
  for (int i = threadIdx.x; i < Hdim; i += 256) { float v = row[i]; ss += v * v; }
#pragma unroll
  for (int m = 16; m >= 1; m >>= 1) ss += __shfl_xor(ss, m, 32);
  __shared__ float warpS[8];
  if ((threadIdx.x & 31) == 0) warpS[threadIdx.x >> 5] = ss;
  __syncthreads();
  __shared__ float rinv;
  if (threadIdx.x < 32) {
    float tot = (threadIdx.x < 8) ? warpS[threadIdx.x] : 0.f;
#pragma unroll
    for (int m = 4; m >= 1; m >>= 1) tot += __shfl_xor(tot, m, 32);
    if (threadIdx.x == 0) rinv = rsqrtf(tot / (float)Hdim + 1e-5f);
  }
  __syncthreads();
  float r = rinv;
  for (int i = threadIdx.x; i < Hdim; i += 256)
    Out[(size_t)n * Hdim + i] = f2bf(row[i] * r * W[i]);
}

// ================= generic WMMA GEMM =================
// C[M,Nn](f32, optional) = A(bf16,row-major) @ B(TB,row-major KxN)
// optional addC (residual), optional accumulate into C, optional bf16 copy.
template <typename TB>
__global__ void __launch_bounds__(256)
gemm_wmma(const __bf16* __restrict__ A, int lda,
          const TB* __restrict__ B, int ldb,
          float* __restrict__ C, int ldc,
          const float* __restrict__ addC,
          __bf16* __restrict__ outBf, int ldo,
          int M, int Nn, int Kk, int accumulate) {
  __shared__ __attribute__((aligned(16))) __bf16 sBT[2][GBN * GSTR];
  int lane = threadIdx.x & 31;
  int wave = threadIdx.x >> 5;
  int wm = wave >> 1, wn = wave & 1;
  int n0 = blockIdx.x * GBN;
  int m0 = blockIdx.y * GBM;
  v8f acc[4] = {{}, {}, {}, {}};
  const __bf16* arow[4];
#pragma unroll
  for (int s = 0; s < 4; ++s)
    arow[s] = A + (size_t)(m0 + wm * 64 + s * 16 + (lane & 15)) * lda;
  int khalf = (lane >> 4) * 8;
  int coll = wn * 16 + (lane & 15);
  int fidx = coll * GSTR + khalf;
  int scol = threadIdx.x & 31;        // 0..31 (cols) -> coalesced global rows
  int skc  = (threadIdx.x >> 5) * 8;  // wave id * 8 -> k chunk
  int sidx = scol * GSTR + skc;
  const TB* bcol = B + n0 + scol;

  // prologue: stage tile 0
  {
    v8bf tmp;
#pragma unroll
    for (int j = 0; j < 8; ++j)
      tmp[j] = f2bf((float)bcol[(size_t)(skc + j) * ldb]);
    *(v8bf*)(&sBT[0][sidx]) = tmp;
  }
  __syncthreads();
  int buf = 0;
  for (int k0 = 0; k0 < Kk; k0 += GBK) {
    // issue next tile's global loads (overlap with WMMAs below)
    v8bf nxt;
    bool have = (k0 + GBK < Kk);
    if (have) {
#pragma unroll
      for (int j = 0; j < 8; ++j)
        nxt[j] = f2bf((float)bcol[(size_t)(k0 + GBK + skc + j) * ldb]);
    }
#pragma unroll
    for (int ks = 0; ks < GBK; ks += 32) {
      v16bf bfrag = frag_from_lds(&sBT[buf][fidx + ks]);  // two ds_load_b128
#pragma unroll
      for (int s = 0; s < 4; ++s) {
        v16bf afrag = load_frag16(arow[s], k0 + ks + khalf);
        acc[s] = wmma_bf16(afrag, bfrag, acc[s]);
      }
    }
    if (have) *(v8bf*)(&sBT[buf ^ 1][sidx]) = nxt;  // one ds_store_b128
    __syncthreads();
    buf ^= 1;
  }
#pragma unroll
  for (int s = 0; s < 4; ++s) {
#pragma unroll
    for (int i = 0; i < 8; ++i) {
      int m = m0 + wm * 64 + s * 16 + (lane >> 4) * 8 + i;
      int n = n0 + coll;
      float v = acc[s][i];
      if (addC) v += addC[(size_t)m * ldc + n];
      if (C) {
        if (accumulate) v += C[(size_t)m * ldc + n];
        C[(size_t)m * ldc + n] = v;
      }
      if (outBf) outBf[(size_t)m * ldo + n] = f2bf(v);
    }
  }
}

// ============= dual-B WMMA GEMM: silu(A@Bg)*(A@Bu)*scale -> bf16 =============
__global__ void __launch_bounds__(256)
gemm_dual_silu(const __bf16* __restrict__ A, int lda,
               const float* __restrict__ Bg, const float* __restrict__ Bu, int ldb,
               const float* __restrict__ rowScale, int scaleStride,
               __bf16* __restrict__ Out, int ldo, int M, int Nn, int Kk) {
  __shared__ __attribute__((aligned(16))) __bf16 sBg[2][GBN * GSTR];
  __shared__ __attribute__((aligned(16))) __bf16 sBu[2][GBN * GSTR];
  int lane = threadIdx.x & 31;
  int wave = threadIdx.x >> 5;
  int wm = wave >> 1, wn = wave & 1;
  int n0 = blockIdx.x * GBN;
  int m0 = blockIdx.y * GBM;
  v8f accg[4] = {{}, {}, {}, {}};
  v8f accu[4] = {{}, {}, {}, {}};
  const __bf16* arow[4];
#pragma unroll
  for (int s = 0; s < 4; ++s)
    arow[s] = A + (size_t)(m0 + wm * 64 + s * 16 + (lane & 15)) * lda;
  int khalf = (lane >> 4) * 8;
  int coll = wn * 16 + (lane & 15);
  int fidx = coll * GSTR + khalf;
  int scol = threadIdx.x & 31;
  int skc  = (threadIdx.x >> 5) * 8;
  int sidx = scol * GSTR + skc;
  const float* bgcol = Bg + n0 + scol;
  const float* bucol = Bu + n0 + scol;

  {
    v8bf tg, tu;
#pragma unroll
    for (int j = 0; j < 8; ++j) {
      size_t gi = (size_t)(skc + j) * ldb;
      tg[j] = f2bf(bgcol[gi]);
      tu[j] = f2bf(bucol[gi]);
    }
    *(v8bf*)(&sBg[0][sidx]) = tg;
    *(v8bf*)(&sBu[0][sidx]) = tu;
  }
  __syncthreads();
  int buf = 0;
  for (int k0 = 0; k0 < Kk; k0 += GBK) {
    v8bf ng, nu;
    bool have = (k0 + GBK < Kk);
    if (have) {
#pragma unroll
      for (int j = 0; j < 8; ++j) {
        size_t gi = (size_t)(k0 + GBK + skc + j) * ldb;
        ng[j] = f2bf(bgcol[gi]);
        nu[j] = f2bf(bucol[gi]);
      }
    }
#pragma unroll
    for (int ks = 0; ks < GBK; ks += 32) {
      v16bf bg = frag_from_lds(&sBg[buf][fidx + ks]);
      v16bf bu = frag_from_lds(&sBu[buf][fidx + ks]);
#pragma unroll
      for (int s = 0; s < 4; ++s) {
        v16bf afrag = load_frag16(arow[s], k0 + ks + khalf);
        accg[s] = wmma_bf16(afrag, bg, accg[s]);
        accu[s] = wmma_bf16(afrag, bu, accu[s]);
      }
    }
    if (have) {
      *(v8bf*)(&sBg[buf ^ 1][sidx]) = ng;
      *(v8bf*)(&sBu[buf ^ 1][sidx]) = nu;
    }
    __syncthreads();
    buf ^= 1;
  }
#pragma unroll
  for (int s = 0; s < 4; ++s) {
#pragma unroll
    for (int i = 0; i < 8; ++i) {
      int m = m0 + wm * 64 + s * 16 + (lane >> 4) * 8 + i;
      int n = n0 + coll;
      float g = accg[s][i], u = accu[s][i];
      float act = (g / (1.f + expf(-g))) * u;
      if (rowScale) act *= rowScale[(size_t)m * scaleStride];
      Out[(size_t)m * ldo + n] = f2bf(act);
    }
  }
}

// ================= MRoPE tables =================
__global__ void __launch_bounds__(64)
mrope_tables(const int* __restrict__ positions, float* __restrict__ cosT,
             float* __restrict__ sinT) {
  int n = blockIdx.x, j = threadIdx.x;  // j in [0,64)
  int row = (j < HWCUT) ? (1 + (j & 1)) : 0;
  float inv = expf(-logf(500000.0f) * (2.0f * (float)j / (float)HD));
  float f = (float)positions[row * NTOK + n] * inv;
  cosT[n * HALF + j] = cosf(f);
  sinT[n * HALF + j] = sinf(f);
}

// ============ RoPE rotate + bf16 pack of Q,K,V ============
__global__ void __launch_bounds__(512)
rope_apply(const float* __restrict__ qkv, const float* __restrict__ cosT,
           const float* __restrict__ sinT, __bf16* __restrict__ qb,
           __bf16* __restrict__ kb, __bf16* __restrict__ vb) {
  int n = blockIdx.x, t = threadIdx.x;
  const float* base = qkv + (size_t)n * QKVW;
  {
    int h = t >> 6, j = t & 63;
    float c = cosT[n * HALF + j], s = sinT[n * HALF + j];
    float x1 = base[h * HD + 2 * j], x2 = base[h * HD + 2 * j + 1];
    size_t o = ((size_t)n * NH + h) * HD + 2 * j;
    qb[o]     = f2bf(x1 * c - x2 * s);
    qb[o + 1] = f2bf(x2 * c + x1 * s);
  }
  if (t < 128) {
    int h = t >> 6, j = t & 63;
    float c = cosT[n * HALF + j], s = sinT[n * HALF + j];
    const float* kbase = base + NH * HD;
    float x1 = kbase[h * HD + 2 * j], x2 = kbase[h * HD + 2 * j + 1];
    size_t o = ((size_t)n * NKV + h) * HD + 2 * j;
    kb[o]     = f2bf(x1 * c - x2 * s);
    kb[o + 1] = f2bf(x2 * c + x1 * s);
  }
  if (t < 256) vb[(size_t)n * (NKV * HD) + t] = f2bf(base[NH * HD + NKV * HD + t]);
}

// ============ attention scores: S[h,q,kk] = scale * q.k, causal ============
__global__ void __launch_bounds__(256)
attn_scores(const __bf16* __restrict__ q, const __bf16* __restrict__ k,
            float* __restrict__ S, float scale) {
  int h  = blockIdx.z;
  int n0 = blockIdx.x * STN;  // key tile
  int m0 = blockIdx.y * STM;  // query tile
  int lane = threadIdx.x & 31;
  int wave = threadIdx.x >> 5;
  int wm = wave >> 2, wn = wave & 3;
  float* Sh = S + (size_t)h * NTOK * NTOK;
  if (n0 >= m0 + STM) {  // fully causal-masked tile
#pragma unroll
    for (int i = 0; i < 8; ++i) {
      int m = m0 + wm * 16 + (lane >> 4) * 8 + i;
      int n = n0 + wn * 16 + (lane & 15);
      Sh[(size_t)m * NTOK + n] = -3.4028235e38f;
    }
    return;
  }
  int kv = h >> 2;  // NH/NKV = 4
  v8f acc = {};
  int qtok = m0 + wm * 16 + (lane & 15);
  int ktok = n0 + wn * 16 + (lane & 15);
  const __bf16* qrow = q + ((size_t)qtok * NH + h) * HD;
  const __bf16* krow = k + ((size_t)ktok * NKV + kv) * HD;
  int khalf = (lane >> 4) * 8;
#pragma unroll
  for (int k0 = 0; k0 < HD; k0 += STK) {
    v16bf af = load_frag16(qrow, k0 + khalf);
    v16bf bf = load_frag16(krow, k0 + khalf);  // B[kk][col] = K[col][kk]
    acc = wmma_bf16(af, bf, acc);
  }
#pragma unroll
  for (int i = 0; i < 8; ++i) {
    int m = m0 + wm * 16 + (lane >> 4) * 8 + i;
    int n = n0 + wn * 16 + (lane & 15);
    float v = (n <= m) ? acc[i] * scale : -3.4028235e38f;
    Sh[(size_t)m * NTOK + n] = v;
  }
}

// ============ row softmax -> bf16 P ============
__global__ void __launch_bounds__(256)
softmax_row_bf16(const float* __restrict__ S, __bf16* __restrict__ P, int L) {
  size_t r = blockIdx.x;
  const float* row = S + r * L;
  __shared__ float warpS[8];
  __shared__ float bcast;
  float mx = -3.4028235e38f;
  for (int i = threadIdx.x; i < L; i += 256) mx = fmaxf(mx, row[i]);
#pragma unroll
  for (int m = 16; m >= 1; m >>= 1) mx = fmaxf(mx, __shfl_xor(mx, m, 32));
  if ((threadIdx.x & 31) == 0) warpS[threadIdx.x >> 5] = mx;
  __syncthreads();
  if (threadIdx.x < 32) {
    float t = (threadIdx.x < 8) ? warpS[threadIdx.x] : -3.4028235e38f;
#pragma unroll
    for (int m = 4; m >= 1; m >>= 1) t = fmaxf(t, __shfl_xor(t, m, 32));
    if (threadIdx.x == 0) bcast = t;
  }
  __syncthreads();
  float rmax = bcast;
  float sum = 0.f;
  for (int i = threadIdx.x; i < L; i += 256) sum += expf(row[i] - rmax);
#pragma unroll
  for (int m = 16; m >= 1; m >>= 1) sum += __shfl_xor(sum, m, 32);
  if ((threadIdx.x & 31) == 0) warpS[threadIdx.x >> 5] = sum;
  __syncthreads();
  if (threadIdx.x < 32) {
    float t = (threadIdx.x < 8) ? warpS[threadIdx.x] : 0.f;
#pragma unroll
    for (int m = 4; m >= 1; m >>= 1) t += __shfl_xor(t, m, 32);
    if (threadIdx.x == 0) bcast = t;
  }
  __syncthreads();
  float inv = 1.f / bcast;
  for (int i = threadIdx.x; i < L; i += 256)
    P[r * L + i] = f2bf(expf(row[i] - rmax) * inv);
}

// ============ gating: softmax(16) -> top-6 -> renorm -> mask ============
__global__ void __launch_bounds__(32)
gate_topk(const __bf16* __restrict__ X, const float* __restrict__ tg,
          const float* __restrict__ ig, const unsigned char* __restrict__ vis,
          float* __restrict__ routing) {
  int n = blockIdx.x;
  int t = threadIdx.x;
  int route = t >> 4, e = t & 15;
  const float* G = route ? ig : tg;
  float acc = 0.f;
  for (int kk = 0; kk < HDIM; ++kk)
    acc += (float)X[(size_t)n * HDIM + kk] * G[(size_t)kk * NEXP + e];
  float mx = acc;
#pragma unroll
  for (int m = 8; m >= 1; m >>= 1) mx = fmaxf(mx, __shfl_xor(mx, m, 32));
  float ex = expf(acc - mx);
  float sm = ex;
#pragma unroll
  for (int m = 8; m >= 1; m >>= 1) sm += __shfl_xor(sm, m, 32);
  float prob = ex / sm;
  __shared__ float probs[32];
  probs[t] = prob;
  __syncthreads();
  if ((t & 15) == 0) {
    bool visTok = vis[n] != 0;
    bool active = ((route == 1) == visTok);
    float* out = routing + ((size_t)route * NTOK + n) * NEXP;
    float w[NEXP];
    int chosen[NEXP];
    float keep[NEXP];
    for (int i = 0; i < NEXP; ++i) { w[i] = probs[route * 16 + i]; chosen[i] = 0; keep[i] = 0.f; }
    float ssum = 0.f;
    for (int kk = 0; kk < KTOP; ++kk) {
      int best = -1; float bv = -1.f;
      for (int i = 0; i < NEXP; ++i)
        if (!chosen[i] && w[i] > bv) { bv = w[i]; best = i; }
      chosen[best] = 1; keep[best] = bv; ssum += bv;
    }
    float invs = 1.f / ssum;
    for (int i = 0; i < NEXP; ++i) out[i] = active ? keep[i] * invs : 0.f;
  }
}

// ================= host-side launch =================
static inline char* wsoff(void* base, size_t& off, size_t bytes) {
  char* p = (char*)base + off;
  off += (bytes + 255) & ~((size_t)255);
  return p;
}

extern "C" void kernel_launch(void* const* d_in, const int* in_sizes, int n_in,
                              void* d_out, int out_size, void* d_ws, size_t ws_size,
                              hipStream_t stream) {
  (void)in_sizes; (void)n_in; (void)out_size; (void)ws_size;
  const float* hidden  = (const float*)d_in[0];
  const float* w_ln_in = (const float*)d_in[1];
  const float* w_ln_po = (const float*)d_in[2];
  const float* w_qkv   = (const float*)d_in[3];
  const float* w_o     = (const float*)d_in[4];
  const float* tgate   = (const float*)d_in[5];
  const float* igate   = (const float*)d_in[6];
  const float* tw_g    = (const float*)d_in[7];
  const float* tw_u    = (const float*)d_in[8];
  const float* tw_d    = (const float*)d_in[9];
  const float* iw_g    = (const float*)d_in[10];
  const float* iw_u    = (const float*)d_in[11];
  const float* iw_d    = (const float*)d_in[12];
  const float* sw_g    = (const float*)d_in[13];
  const float* sw_u    = (const float*)d_in[14];
  const float* sw_d    = (const float*)d_in[15];
  const int*   positions = (const int*)d_in[16];
  const unsigned char* vis = (const unsigned char*)d_in[17];
  float* out = (float*)d_out;

  size_t off = 0;
  __bf16* xnorm   = (__bf16*)wsoff(d_ws, off, (size_t)NTOK * HDIM * 2);
  float*  qkv     = (float*) wsoff(d_ws, off, (size_t)NTOK * QKVW * 4);
  float*  cosT    = (float*) wsoff(d_ws, off, (size_t)NTOK * HALF * 4);
  float*  sinT    = (float*) wsoff(d_ws, off, (size_t)NTOK * HALF * 4);
  __bf16* qb      = (__bf16*)wsoff(d_ws, off, (size_t)NTOK * NH * HD * 2);
  __bf16* kb      = (__bf16*)wsoff(d_ws, off, (size_t)NTOK * NKV * HD * 2);
  __bf16* vb      = (__bf16*)wsoff(d_ws, off, (size_t)NTOK * NKV * HD * 2);
  float*  S       = (float*) wsoff(d_ws, off, (size_t)NH * NTOK * NTOK * 4);
  __bf16* P       = (__bf16*)wsoff(d_ws, off, (size_t)NH * NTOK * NTOK * 2);
  __bf16* attnb   = (__bf16*)wsoff(d_ws, off, (size_t)NTOK * HDIM * 2);
  float*  hbuf    = (float*) wsoff(d_ws, off, (size_t)NTOK * HDIM * 4);
  __bf16* x2b     = (__bf16*)wsoff(d_ws, off, (size_t)NTOK * HDIM * 2);
  __bf16* actsh   = (__bf16*)wsoff(d_ws, off, (size_t)NTOK * SIDIM * 2);
  float*  routing = (float*) wsoff(d_ws, off, (size_t)2 * NTOK * NEXP * 4);
  __bf16* acte    = (__bf16*)wsoff(d_ws, off, (size_t)NTOK * IDIM * 2);

  // 1) rms(hidden, w_ln_in) -> bf16
  rmsnorm_bf16<<<NTOK, 256, 0, stream>>>(hidden, w_ln_in, xnorm, HDIM);
  // 2) qkv = xnorm @ w_qkv
  gemm_wmma<float><<<dim3(QKVW / GBN, NTOK / GBM), 256, 0, stream>>>(
      xnorm, HDIM, w_qkv, QKVW, qkv, QKVW, nullptr, nullptr, 0,
      NTOK, QKVW, HDIM, 0);
  // 3) MRoPE tables + rotate
  mrope_tables<<<NTOK, 64, 0, stream>>>(positions, cosT, sinT);
  rope_apply<<<NTOK, 512, 0, stream>>>(qkv, cosT, sinT, qb, kb, vb);
  // 4) attention scores (causal, scaled)
  attn_scores<<<dim3(NTOK / STN, NTOK / STM, NH), 256, 0, stream>>>(
      qb, kb, S, 0.08838834764831845f);
  // 5) softmax rows -> P bf16
  softmax_row_bf16<<<NH * NTOK, 256, 0, stream>>>(S, P, NTOK);
  // 6) O = P @ V per head -> attn bf16 (N x H)
  for (int h = 0; h < NH; ++h) {
    int kv = h >> 2;
    gemm_wmma<__bf16><<<dim3(HD / GBN, NTOK / GBM), 256, 0, stream>>>(
        P + (size_t)h * NTOK * NTOK, NTOK, vb + (size_t)kv * HD, NKV * HD,
        nullptr, 0, nullptr, attnb + (size_t)h * HD, HDIM,
        NTOK, HD, NTOK, 0);
  }
  // 7) h = hidden + attn @ w_o
  gemm_wmma<float><<<dim3(HDIM / GBN, NTOK / GBM), 256, 0, stream>>>(
      attnb, HDIM, w_o, HDIM, hbuf, HDIM, hidden, nullptr, 0,
      NTOK, HDIM, HDIM, 0);
  // 8) x2 = rms(h, w_ln_post)
  rmsnorm_bf16<<<NTOK, 256, 0, stream>>>(hbuf, w_ln_po, x2b, HDIM);
  // 9) shared expert: act = silu(x2@sw_g)*(x2@sw_u)
  gemm_dual_silu<<<dim3(SIDIM / GBN, NTOK / GBM), 256, 0, stream>>>(
      x2b, HDIM, sw_g, sw_u, SIDIM, nullptr, 0, actsh, SIDIM,
      NTOK, SIDIM, HDIM);
  // 10) out = h + act @ sw_d   (initializes d_out fully)
  gemm_wmma<float><<<dim3(HDIM / GBN, NTOK / GBM), 256, 0, stream>>>(
      actsh, SIDIM, sw_d, HDIM, out, HDIM, hbuf, nullptr, 0,
      NTOK, HDIM, SIDIM, 0);
  // 11) routing weights (softmax -> top-6 -> renorm -> mask)
  gate_topk<<<NTOK, 32, 0, stream>>>(x2b, tgate, igate, vis, routing);
  // 12) routed experts: for each (route, expert): act_e then out += act_e @ wd
  for (int route = 0; route < 2; ++route) {
    const float* wgb = route ? iw_g : tw_g;
    const float* wub = route ? iw_u : tw_u;
    const float* wdb = route ? iw_d : tw_d;
    for (int e = 0; e < NEXP; ++e) {
      const float* wg = wgb + (size_t)e * HDIM * IDIM;
      const float* wu = wub + (size_t)e * HDIM * IDIM;
      const float* wd = wdb + (size_t)e * IDIM * HDIM;
      const float* rs = routing + (size_t)route * NTOK * NEXP + e;
      gemm_dual_silu<<<dim3(IDIM / GBN, NTOK / GBM), 256, 0, stream>>>(
          x2b, HDIM, wg, wu, IDIM, rs, NEXP, acte, IDIM,
          NTOK, IDIM, HDIM);
      gemm_wmma<float><<<dim3(HDIM / GBN, NTOK / GBM), 256, 0, stream>>>(
          acte, IDIM, wd, HDIM, out, HDIM, nullptr, nullptr, 0,
          NTOK, HDIM, IDIM, 1 /*accumulate*/);
    }
  }
}
